// GhostGRUCell_8426725835022
// MI455X (gfx1250) — matrix operand order repaired and checked by reading, not compiled, not run on hardware
//
#include <hip/hip_runtime.h>

// GhostGRU for MI455X (gfx1250), single persistent kernel.
// The recurrence is row-local in batch -> each workgroup owns 16 batch rows
// and runs all T=512 steps with only workgroup barriers. State lives in LDS
// (120 KB of the WGP's 320 KB). A-operands come from LDS (ds_load_b128),
// B-operands are L2-resident bf16 weights (global_load_b128), matrix math is
// v_wmma_f32_16x16x32_bf16 with f32 accumulate.
//
// Anti-spill / anti-LICM: an opaque (asm-laundered) integer offset is added
// to the weight pointers each timestep, so weight loads cannot be hoisted
// out of the t-loop -- but the pointer chain stays intact so loads remain in
// the GLOBAL address space (not flat). K loops use #pragma unroll 2.
// Nonlinearities use v_exp_f32 + v_rcp_f32 (branch-free, ~4 VALU ops) since
// the bf16 GEMM makes correctly-rounded f32 division pointless.

typedef __attribute__((ext_vector_type(16))) __bf16 v16bf;
typedef __attribute__((ext_vector_type(8)))  __bf16 v8bf;
typedef __attribute__((ext_vector_type(8)))  float  v8f;

#define B_    128
#define T_    512
#define F_    256
#define H_    1024
#define GH_   512
#define IN_   512
#define KBIG  1280   // F_ + H_
#define NGATE 1024   // 2 * IN_

// ---- float -> bf16 (round to nearest even), pure integer ops ----
static __device__ __forceinline__ __bf16 f2bf(float f) {
  unsigned int u = __builtin_bit_cast(unsigned int, f);
  u = (u + 0x7FFFu + ((u >> 16) & 1u)) >> 16;
  unsigned short s = (unsigned short)u;
  return __builtin_bit_cast(__bf16, s);
}

// ---- fast branch-free nonlinearities (v_exp_f32 + v_rcp_f32) ----
// Saturate correctly: exp overflow -> inf -> rcp -> 0.
static __device__ __forceinline__ float fast_sigmoid(float x) {
  return __builtin_amdgcn_rcpf(1.0f + __expf(-x));
}
static __device__ __forceinline__ float fast_tanh(float x) {
  return 1.0f - 2.0f * __builtin_amdgcn_rcpf(1.0f + __expf(2.0f * x));
}

// =====================================================================
// K-segment helpers for a 16(M) x {32,16}(N) wave tile.
//   ap : lane's A row data at segment start (+kHalf applied) -- LDS
//   wl : W + n0*ldk + segStart + kBH (row-major (N,K) bf16 weights)
// A-fragment (16-bit A 16x32, contiguous K order): lane m<16 holds K {0..15},
// lane m+16 holds K {16..31} -> two 16B ds_load_b128 runs.
// B-fragment (32x16): lane holds 16 consecutive K of one column -> one
// aligned 32B contiguous global load (2 x b128).
// =====================================================================
static __device__ __forceinline__ void gemm_seg2(
    const __bf16* ap, const __bf16* __restrict__ wl,
    size_t ldk, int klen, v8f acc[2]) {
#pragma unroll 2
  for (int kk = 0; kk < klen; kk += 32) {
    v8bf r0 = *(const v8bf*)(ap + kk);
    v8bf r1 = *(const v8bf*)(ap + kk + 8);
    v16bf a;
#pragma unroll
    for (int e = 0; e < 8; ++e) { a[e] = r0[e]; a[e + 8] = r1[e]; }
#pragma unroll
    for (int j = 0; j < 2; ++j) {
      v16bf b = *(const v16bf*)(wl + (size_t)j * 16 * ldk + kk);
      acc[j] = __builtin_amdgcn_wmma_f32_16x16x32_bf16(
          false, a, false, b, (short)0, acc[j], false, false);
    }
  }
}

static __device__ __forceinline__ void gemm_seg1(
    const __bf16* ap, const __bf16* __restrict__ wl,
    int klen, v8f& acc) {
#pragma unroll 2
  for (int kk = 0; kk < klen; kk += 32) {
    v8bf r0 = *(const v8bf*)(ap + kk);
    v8bf r1 = *(const v8bf*)(ap + kk + 8);
    v16bf a;
#pragma unroll
    for (int e = 0; e < 8; ++e) { a[e] = r0[e]; a[e + 8] = r1[e]; }
    v16bf b = *(const v16bf*)(wl + kk);
    acc = __builtin_amdgcn_wmma_f32_16x16x32_bf16(
        false, a, false, b, (short)0, acc, false, false);
  }
}

// =====================================================================
// Persistent GhostGRU kernel. grid = 8 blocks (16 batch rows each),
// block = 1024 threads = 32 waves. All T steps inside.
// Per step:
//   gate : wave w -> cols [w*32, w*32+32) of 1024   (2 accums)
//   cand : wave w -> cols [w*16, w*16+16) of 512    (1 accum)
//   ghost: wave w -> cols [w*16, w*16+16) of 512    (1 accum)
// =====================================================================
__global__ __launch_bounds__(1024) void ghostgru_kernel(
    const __bf16* __restrict__ x16,  // (B,T,F) bf16
    const float*  __restrict__ hid,  // (B,1,H) f32
    const __bf16* __restrict__ Wg,   // (1024,1280) bf16
    const __bf16* __restrict__ Wc,   // (512,1280)  bf16
    const __bf16* __restrict__ Wh,   // (512,512)   bf16
    const float*  __restrict__ bg,   // (1024)
    const float*  __restrict__ bc,   // (512)
    const float*  __restrict__ bgh,  // (512)
    float*        __restrict__ out)  // (B,T,H) f32
{
  __shared__ __bf16 hbf[16][H_];    // h bf16: [ghost | state]   32 KB
  __shared__ float  hsf[16][IN_];   // h state f32               32 KB
  __shared__ __bf16 rsm[16][IN_];   // r * h_state, bf16         16 KB
  __shared__ float  ugm[16][IN_];   // u gate f32                32 KB
  __shared__ __bf16 xbf[16][F_];    // x_t staging                8 KB

  const int lane  = threadIdx.x & 31;
  const int wave  = threadIdx.x >> 5;       // 0..31
  const int mBase = blockIdx.x * 16;        // rows owned by this block
  const int l15   = lane & 15;
  const int hi    = lane >> 4;              // 0/1 half-wave
  const int kHalf = hi << 4;                // A: 0 or 16
  const int kBH   = hi << 4;                // B: 0 or 16
  const int mLoc  = hi << 3;                // C/D: local M base (0 or 8)

  // gate tile: cols n0g, n0g+16 ; cand/ghost tile: col nc
  const int n0g = wave * 32 + l15;
  const int nc  = wave * 16 + l15;

  // Weight row pointers and biases (invariant over t).
  const __bf16* wlg = Wg + (size_t)n0g * KBIG + kBH;
  const __bf16* wlc = Wc + (size_t)nc  * KBIG + kBH;
  const __bf16* wlh = Wh + (size_t)nc  * IN_  + kBH;
  const float bg0 = bg[n0g], bg1 = bg[n0g + 16];
  const float bcv = bc[nc];
  const float bhv = bgh[nc];

  // ---- initial state from `hidden` ----
  for (int i = threadIdx.x; i < 16 * H_; i += 1024) {
    int r = i >> 10, c = i & (H_ - 1);
    float v = hid[(size_t)(mBase + r) * H_ + c];
    hbf[r][c] = f2bf(v);
    if (c >= GH_) hsf[r][c - GH_] = v;
  }
  // ---- stage x_0 (16 rows x 256 bf16 = 8 KB, 16B per thread) ----
  if (threadIdx.x < 512) {
    int idx = threadIdx.x * 8;
    int r = idx >> 8, c = idx & (F_ - 1);
    *(v8bf*)&xbf[r][c] = *(const v8bf*)(x16 + ((size_t)(mBase + r) * T_) * F_ + c);
  }
  __syncthreads();

  for (int t = 0; t < T_; ++t) {
    // Opaque zero offset: the compiler cannot prove the weight-load
    // addresses t-invariant, so no LICM/spill -- but the pointer chain is
    // preserved, so loads stay GLOBAL (not flat).
    unsigned off = 0;
    asm volatile("" : "+s"(off));
    const __bf16* wg_t = wlg + off;
    const __bf16* wc_t = wlc + off;
    const __bf16* wh_t = wlh + off;

    // ================= gate phase =================
    {
      v8f acc[2];
#pragma unroll
      for (int i = 0; i < 8; ++i) { acc[0][i] = bg0; acc[1][i] = bg1; }
      gemm_seg2(&xbf[l15][kHalf], wg_t,      KBIG, F_, acc);  // K [0,256): x_t
      gemm_seg2(&hbf[l15][kHalf], wg_t + F_, KBIG, H_, acc);  // K [256,1280): h

#pragma unroll
      for (int j = 0; j < 2; ++j) {
        const int n = n0g + j * 16;
#pragma unroll
        for (int i = 0; i < 8; ++i) {
          const int m = mLoc + i;
          float g = fast_sigmoid(acc[j][i]);
          if (n < IN_) rsm[m][n] = f2bf(g * hsf[m][n]);   // r * h_state
          else         ugm[m][n - IN_] = g;               // u
        }
      }
    }
    __syncthreads();

    // ================= candidate phase =================
    {
      v8f acc;
#pragma unroll
      for (int i = 0; i < 8; ++i) acc[i] = bcv;
      gemm_seg1(&xbf[l15][kHalf], wc_t,             F_,  acc);  // x_t
      gemm_seg1(&hbf[l15][kHalf], wc_t + F_,        GH_, acc);  // h ghost
      gemm_seg1(&rsm[l15][kHalf], wc_t + F_ + GH_,  IN_, acc);  // r*h_state

#pragma unroll
      for (int i = 0; i < 8; ++i) {
        const int m = mLoc + i;
        float c  = fast_tanh(acc[i]);
        float u  = ugm[m][nc];
        float hs = hsf[m][nc];
        float nh = u * hs + (1.0f - u) * c;
        out[((size_t)(mBase + m) * T_ + t) * H_ + GH_ + nc] = nh;
        hsf[m][nc] = nh;
        hbf[m][GH_ + nc] = f2bf(nh);
      }
    }
    __syncthreads();

    // ================= ghost phase =================
    {
      v8f acc;
#pragma unroll
      for (int i = 0; i < 8; ++i) acc[i] = bhv;
      gemm_seg1(&hbf[l15][GH_ + kHalf], wh_t, IN_, acc);        // new_h

#pragma unroll
      for (int i = 0; i < 8; ++i) {
        const int m = mLoc + i;
        float g = fast_tanh(acc[i]);
        out[((size_t)(mBase + m) * T_ + t) * H_ + nc] = g;
        hbf[m][nc] = f2bf(g);
      }
    }
    // stage x_{t+1} (xbf last read in cand phase, before previous barrier)
    if (t + 1 < T_ && threadIdx.x < 512) {
      int idx = threadIdx.x * 8;
      int r = idx >> 8, c = idx & (F_ - 1);
      *(v8bf*)&xbf[r][c] =
          *(const v8bf*)(x16 + ((size_t)(mBase + r) * T_ + (t + 1)) * F_ + c);
    }
    __syncthreads();
  }
}

// ---- f32 -> bf16 bulk convert (weights, x) ----
__global__ void cvt_kernel(const float* __restrict__ src,
                           __bf16* __restrict__ dst, int n) {
  int i = blockIdx.x * blockDim.x + threadIdx.x;
  if (i < n) dst[i] = f2bf(src[i]);
}

// ---- gather final hidden: h_final[b, :] = out[b, T-1, :] ----
__global__ void final_kernel(const float* __restrict__ out,
                             float* __restrict__ hfin) {
  int i = blockIdx.x * blockDim.x + threadIdx.x;
  if (i < B_ * H_) {
    int b = i >> 10, c = i & (H_ - 1);
    hfin[i] = out[((size_t)b * T_ + (T_ - 1)) * H_ + c];
  }
}

extern "C" void kernel_launch(void* const* d_in, const int* in_sizes, int n_in,
                              void* d_out, int out_size, void* d_ws, size_t ws_size,
                              hipStream_t stream) {
  (void)in_sizes; (void)n_in; (void)out_size; (void)ws_size;

  const float* x   = (const float*)d_in[0];  // (B,T,F)
  const float* hid = (const float*)d_in[1];  // (B,1,H)
  const float* wg  = (const float*)d_in[2];  // (1024,1280)
  const float* wc  = (const float*)d_in[3];  // (512,1280)
  const float* wh  = (const float*)d_in[4];  // (512,512)
  const float* bg  = (const float*)d_in[5];  // (1024)
  const float* bc  = (const float*)d_in[6];  // (512)
  const float* bgh = (const float*)d_in[7];  // (512)

  float* out  = (float*)d_out;                    // (B,T,H) flat
  float* hfin = out + (size_t)B_ * T_ * H_;       // (B,1,H) flat

  // Workspace partition (all 32B-aligned): ~38 MB
  char* ws = (char*)d_ws;
  __bf16* Wg16 = (__bf16*)ws;  ws += (size_t)NGATE * KBIG * 2;
  __bf16* Wc16 = (__bf16*)ws;  ws += (size_t)IN_   * KBIG * 2;
  __bf16* Wh16 = (__bf16*)ws;  ws += (size_t)GH_   * IN_  * 2;
  __bf16* x16  = (__bf16*)ws;  ws += (size_t)B_ * T_ * F_ * 2;

  // One-time conversions: weights (4.5 MB bf16, L2-resident for the whole
  // run) and x (~170 MB HBM traffic ~ 7 us at 23.3 TB/s). This removes all
  // f32->bf16 VALU work from the 512-step sequential critical path.
  cvt_kernel<<<(NGATE * KBIG + 255) / 256, 256, 0, stream>>>(wg, Wg16, NGATE * KBIG);
  cvt_kernel<<<(IN_ * KBIG + 255) / 256, 256, 0, stream>>>(wc, Wc16, IN_ * KBIG);
  cvt_kernel<<<(GH_ * IN_ + 255) / 256, 256, 0, stream>>>(wh, Wh16, GH_ * IN_);
  cvt_kernel<<<(B_ * T_ * F_ + 255) / 256, 256, 0, stream>>>(x, x16, B_ * T_ * F_);

  // One persistent kernel runs the whole recurrence: 8 independent blocks
  // (16 batch rows each), no inter-block communication.
  ghostgru_kernel<<<8, 1024, 0, stream>>>(x16, hid, Wg16, Wc16, Wh16,
                                          bg, bc, bgh, out);

  final_kernel<<<(B_ * H_ + 255) / 256, 256, 0, stream>>>(out, hfin);
}
